// VectorQuantizer_26645977104470
// MI455X (gfx1250) — matrix operand order repaired and checked by reading, not compiled, not run on hardware
//
#include <hip/hip_runtime.h>
#include <hip/hip_bf16.h>
#include <math.h>

typedef __attribute__((ext_vector_type(16))) __bf16 v16bf;
typedef __attribute__((ext_vector_type(8)))  float  v8f;

#define LATENT_DIM 256
#define KCODES     8192
#define NROWS      16384
#define TILE_CODES 64
#define NTILES     (KCODES / TILE_CODES)      // 128
#define ROW_STRIDE (LATENT_DIM + 16)          // 272 bf16 -> 544B row, spreads LDS banks
#define TILE_HALFS (TILE_CODES * ROW_STRIDE)  // bf16 elements per LDS buffer
#define ROWS_PER_WAVE 32                      // 2 M-subtiles per wave: B reused 2x per ds load
#define WAVES_GEMM 4
#define THREADS_GEMM (WAVES_GEMM * 32)        // 128
#define BLK_ROWS   (WAVES_GEMM * ROWS_PER_WAVE) // 128

// d_out layout (floats)
#define SYM_OFF  0
#define Q_OFF    16384
#define ST_OFF   (16384 + NROWS * LATENT_DIM)
#define LOSS_OFF (ST_OFF + NROWS * LATENT_DIM)

__device__ __forceinline__ unsigned short f32_to_bf16(float f) {
  unsigned u = __float_as_uint(f);
  u += 0x7FFFu + ((u >> 16) & 1u);            // round-to-nearest-even
  return (unsigned short)(u >> 16);
}

// ---------------- kernel 1: codebook f32 -> bf16, zero histogram ----------------
__global__ void vq_prep(const float* __restrict__ cb, unsigned short* __restrict__ cbh,
                        float* __restrict__ hist) {
  int gid = blockIdx.x * blockDim.x + threadIdx.x;    // 0 .. 2,097,151
  cbh[gid] = f32_to_bf16(cb[gid]);
  if (gid < KCODES) hist[gid] = 0.0f;
}

// ---------------- kernel 2: exact f32 ||c||^2 per code (1 wave / code) ----------------
__global__ void vq_cnorm(const float* __restrict__ cb, float* __restrict__ cnorm) {
  int wave = threadIdx.x >> 5, lane = threadIdx.x & 31;
  int code = blockIdx.x * 8 + wave;
  const float* row = cb + code * LATENT_DIM;
  float s = 0.0f;
  #pragma unroll
  for (int i = 0; i < LATENT_DIM / 32; ++i) { float v = row[lane + i * 32]; s += v * v; }
  #pragma unroll
  for (int m = 16; m; m >>= 1) s += __shfl_xor(s, m, 32);
  if (lane == 0) cnorm[code] = s;
}

// ---------------- kernel 3: fused bf16-WMMA GEMM + running argmin ----------------
__device__ __forceinline__ void vq_issue_tile(unsigned ldsBufBase, unsigned long long cbBase,
                                              int codebase, int tid) {
  // 64 codes x 512B -> LDS in 16B chunks; 16 async instrs per thread (per wave: ASYNCcnt += 16)
  #pragma unroll
  for (int i = 0; i < 16; ++i) {
    int chunk = tid + i * THREADS_GEMM;       // 0..2047
    int code  = chunk >> 5;
    int off16 = chunk & 31;
    unsigned ldsOff = ldsBufBase + (unsigned)(code * ROW_STRIDE * 2 + off16 * 16);
    unsigned gOff   = (unsigned)(((codebase + code) * LATENT_DIM) * 2 + off16 * 16);
    asm volatile("global_load_async_to_lds_b128 %0, %1, %2 offset:0"
                 :: "v"(ldsOff), "v"(gOff), "s"(cbBase) : "memory");
  }
}

__global__ void __launch_bounds__(THREADS_GEMM)
vq_gemm_argmin(const float* __restrict__ latents,
               const unsigned short* __restrict__ cbh,
               const float* __restrict__ cnorm,
               int* __restrict__ symidx) {
  extern __shared__ unsigned short tileB[];   // 2 buffers x 64 codes x 272 bf16 (~69.6 KB)

  const int lane    = threadIdx.x & 31;
  const int wave    = threadIdx.x >> 5;
  const int col16   = lane & 15;
  const int rowbase = blockIdx.x * BLK_ROWS + wave * ROWS_PER_WAVE;
  const int khalf8  = (lane < 16) ? 0 : 16;   // A: lane half picks K 16-block
  const int khalf16 = (lane < 16) ? 0 : 16;   // B: lane half picks K 16-block

  // A fragments: 2 M-subtiles x 256 K in bf16, resident in VGPRs (128 VGPRs)
  v16bf afrag[2][8];
  #pragma unroll
  for (int s = 0; s < 2; ++s) {
    const int myrow = rowbase + s * 16 + col16;
    #pragma unroll
    for (int kk = 0; kk < 8; ++kk) {
      union { v16bf v; unsigned short h[16]; } au;
      const float* ap = latents + myrow * LATENT_DIM + kk * 32 + khalf8;
      union { float4 v; float f[4]; } f0, f1, f2, f3;
      f0.v = *(const float4*)(ap);       f1.v = *(const float4*)(ap + 4);
      f2.v = *(const float4*)(ap + 8);   f3.v = *(const float4*)(ap + 12);
      #pragma unroll
      for (int e = 0; e < 4; ++e) {
        au.h[e]      = f32_to_bf16(f0.f[e]);
        au.h[4 + e]  = f32_to_bf16(f1.f[e]);
        au.h[8 + e]  = f32_to_bf16(f2.f[e]);
        au.h[12 + e] = f32_to_bf16(f3.f[e]);
      }
      afrag[s][kk] = au.v;
    }
  }

  float minv[2][8]; int mini[2][8];
  #pragma unroll
  for (int s = 0; s < 2; ++s)
    #pragma unroll
    for (int r = 0; r < 8; ++r) { minv[s][r] = 3.4e38f; mini[s][r] = 0; }

  const unsigned bufBase0 = (unsigned)(size_t)(&tileB[0]); // LDS aperture: low 32 bits = offset
  const unsigned bufBase1 = bufBase0 + TILE_HALFS * 2;
  const unsigned long long cbBase = (unsigned long long)(size_t)cbh;

  vq_issue_tile(bufBase0, cbBase, 0, threadIdx.x);         // prefetch tile 0

  for (int t = 0; t < NTILES; ++t) {
    const int codebase = t * TILE_CODES;
    if (t + 1 < NTILES) {
      vq_issue_tile(((t + 1) & 1) ? bufBase1 : bufBase0, cbBase,
                    (t + 1) * TILE_CODES, threadIdx.x);
      asm volatile("s_wait_asynccnt 0x10" ::: "memory");   // tile t resident (in-order), t+1 in flight
    } else {
      asm volatile("s_wait_asynccnt 0x0" ::: "memory");
    }
    __syncthreads();

    const unsigned short* buf = tileB + (size_t)(t & 1) * TILE_HALFS;

    float cn[4];                                           // hoisted ||c||^2 loads, overlap WMMAs
    #pragma unroll
    for (int sub = 0; sub < 4; ++sub) cn[sub] = cnorm[codebase + sub * 16 + col16];

    #pragma unroll
    for (int sub = 0; sub < 4; ++sub) {
      const int col  = sub * 16 + col16;                   // C-layout: N = lane%16
      const int code = codebase + col;

      v16bf bfrag[8];                                      // preload -> DS loads pipeline
      #pragma unroll
      for (int kk = 0; kk < 8; ++kk)
        bfrag[kk] = *(const v16bf*)(&buf[col * ROW_STRIDE + kk * 32 + khalf16]);

      v8f acc0 = {}, acc1 = {};
      #pragma unroll
      for (int kk = 0; kk < 8; ++kk) {                     // each B fragment feeds 2 WMMAs
        acc0 = __builtin_amdgcn_wmma_f32_16x16x32_bf16(false, afrag[0][kk], false, bfrag[kk],
                                                       (short)0, acc0, false, false);
        acc1 = __builtin_amdgcn_wmma_f32_16x16x32_bf16(false, afrag[1][kk], false, bfrag[kk],
                                                       (short)0, acc1, false, false);
      }
      union { v8f v; float f[8]; } a0, a1; a0.v = acc0; a1.v = acc1;
      #pragma unroll
      for (int r = 0; r < 8; ++r) {                        // slot r -> row r (lo half) / 8+r (hi)
        float d0 = cn[sub] - 2.0f * a0.f[r];
        bool b0 = d0 < minv[0][r];
        mini[0][r] = b0 ? code : mini[0][r];
        minv[0][r] = b0 ? d0 : minv[0][r];
        float d1 = cn[sub] - 2.0f * a1.f[r];
        bool b1 = d1 < minv[1][r];
        mini[1][r] = b1 ? code : mini[1][r];
        minv[1][r] = b1 ? d1 : minv[1][r];
      }
    }
    __syncthreads();                                       // reads done before buf reuse at t+2
  }

  // argmin across the 16 lanes holding each row's columns (ties -> lowest code id)
  #pragma unroll
  for (int m = 1; m <= 8; m <<= 1) {
    #pragma unroll
    for (int s = 0; s < 2; ++s)
      #pragma unroll
      for (int r = 0; r < 8; ++r) {
        float ov = __shfl_xor(minv[s][r], m, 32);
        int   oi = __shfl_xor(mini[s][r], m, 32);
        bool take = (ov < minv[s][r]) || (ov == minv[s][r] && oi < mini[s][r]);
        minv[s][r] = take ? ov : minv[s][r];
        mini[s][r] = take ? oi : mini[s][r];
      }
  }
  if (col16 == 0) {
    #pragma unroll
    for (int s = 0; s < 2; ++s) {
      int mrow = rowbase + s * 16 + ((lane < 16) ? 0 : 8);
      #pragma unroll
      for (int r = 0; r < 8; ++r) symidx[mrow + r] = mini[s][r];
    }
  }
}

// ---------------- kernel 4: gather + MSE partials + masked histogram ----------------
__global__ void vq_gather(const float* __restrict__ latents, const float* __restrict__ cb,
                          const int* __restrict__ symidx, const float* __restrict__ mask,
                          float* __restrict__ hist, float* __restrict__ partial,
                          float* __restrict__ out) {
  const int row = blockIdx.x;
  const int t   = threadIdx.x;
  const int sym = symidx[row];
  const float l = latents[row * LATENT_DIM + t];
  const float q = cb[sym * LATENT_DIM + t];
  out[Q_OFF  + row * LATENT_DIM + t] = q;
  out[ST_OFF + row * LATENT_DIM + t] = l + (q - l);   // straight-through forward value
  float d = l - q; d *= d;

  __shared__ float red[8];
  #pragma unroll
  for (int m = 16; m; m >>= 1) d += __shfl_xor(d, m, 32);
  if ((t & 31) == 0) red[t >> 5] = d;
  __syncthreads();
  if (t == 0) {
    float s = 0.0f;
    #pragma unroll
    for (int i = 0; i < 8; ++i) s += red[i];
    partial[row] = s;                                  // deterministic reduction path
    atomicAdd(&hist[sym], mask[row]);                  // segment_sum of patch_mask
    out[SYM_OFF + row] = (float)sym;
  }
}

// ---------------- kernel 5: losses + perplexity ----------------
__device__ float vq_block_reduce(float v, float* red) {
  #pragma unroll
  for (int m = 16; m; m >>= 1) v += __shfl_xor(v, m, 32);
  if ((threadIdx.x & 31) == 0) red[threadIdx.x >> 5] = v;
  __syncthreads();
  float r = 0.0f;
  if (threadIdx.x == 0)
    for (int i = 0; i < 8; ++i) r += red[i];
  __syncthreads();
  return r;   // valid on thread 0
}

__global__ void vq_final(const float* __restrict__ mask, const float* __restrict__ hist,
                         const float* __restrict__ partial, float* __restrict__ out) {
  __shared__ float red[8];
  __shared__ float sh_denom;
  const int t = threadIdx.x;

  float ms = 0.0f;
  for (int i = t; i < NROWS; i += 256) ms += mask[i];
  ms = vq_block_reduce(ms, red);
  if (t == 0) sh_denom = fmaxf(ms, 1.0f);
  __syncthreads();
  const float denom = sh_denom;

  float se = 0.0f;
  for (int i = t; i < NROWS; i += 256) se += partial[i];
  se = vq_block_reduce(se, red);

  float e = 0.0f;
  for (int i = t; i < KCODES; i += 256) {
    float p = hist[i] / denom;
    e += p * logf(p + 1e-8f);
  }
  e = vq_block_reduce(e, red);

  if (t == 0) {
    float mse = se / (float)(NROWS * LATENT_DIM);
    out[LOSS_OFF + 0] = 0.25f * mse;   // commitment
    out[LOSS_OFF + 1] = mse;           // codebook
    out[LOSS_OFF + 2] = expf(-e);      // perplexity
  }
}

extern "C" void kernel_launch(void* const* d_in, const int* in_sizes, int n_in,
                              void* d_out, int out_size, void* d_ws, size_t ws_size,
                              hipStream_t stream) {
  (void)in_sizes; (void)n_in; (void)out_size; (void)ws_size;
  const float* latents = (const float*)d_in[0];   // [16,1024,256]
  const float* mask    = (const float*)d_in[1];   // [16,1024]
  const float* cb      = (const float*)d_in[2];   // [8192,256]
  float* out = (float*)d_out;

  char* ws = (char*)d_ws;
  unsigned short* cbh   = (unsigned short*)(ws);                 // 4,194,304 B bf16 codebook
  float* cnorm          = (float*)(ws + 4194304);                // 32 KB
  float* hist           = (float*)(ws + 4194304 + 32768);        // 32 KB
  int*   symidx         = (int*)  (ws + 4194304 + 65536);        // 64 KB
  float* partial        = (float*)(ws + 4194304 + 131072);       // 64 KB

  const size_t lds_bytes = 2u * TILE_HALFS * sizeof(unsigned short); // double-buffered tiles

  vq_prep<<<(KCODES * LATENT_DIM) / 256, 256, 0, stream>>>(cb, cbh, hist);
  vq_cnorm<<<KCODES / 8, 256, 0, stream>>>(cb, cnorm);
  vq_gemm_argmin<<<NROWS / BLK_ROWS, THREADS_GEMM, lds_bytes, stream>>>(latents, cbh, cnorm, symidx);
  vq_gather<<<NROWS, 256, 0, stream>>>(latents, cb, symidx, mask, hist, partial, out);
  vq_final<<<1, 256, 0, stream>>>(mask, hist, partial, out);
}
